// LSTM_Dec_v1_40106404610738
// MI455X (gfx1250) — matrix operand order repaired and checked by reading, not compiled, not run on hardware
//
#include <hip/hip_runtime.h>
#include <hip/hip_bf16.h>

typedef __attribute__((ext_vector_type(16))) __bf16 v16bf;
typedef __attribute__((ext_vector_type(8)))  __bf16 v8bf;
typedef __attribute__((ext_vector_type(8)))  float  v8f;

#define SEQ    256
#define HID    1024
#define NBATCH 256
#define GATE4  4096

// ---------------- WMMA fragment helpers (CDNA5 layouts, wave32) ----------------
// A (16x32 bf16): lane l -> row (l&15); hi=(l>>4): elems 0..7 = K[hi*16 .. +7],
//                 elems 8..15 = K[hi*16+8 .. +7]   (two 16B contiguous loads)
static __device__ __forceinline__ v16bf load_a_frag(const __bf16* __restrict__ Arow,
                                                    int k0, int hi) {
  v8bf lo = *(const v8bf*)(Arow + k0 + hi * 16);
  v8bf hv = *(const v8bf*)(Arow + k0 + hi * 16 + 8);
  v16bf a;
#pragma unroll
  for (int i = 0; i < 8; ++i) { a[i] = lo[i]; a[i + 8] = hv[i]; }
  return a;
}

// B (32x16 bf16) for X @ W^T with W row-major [N,K]: lane l -> col n=(l&15),
// holds B[k0 + (l>>4)*16 .. +15, n] = W[n, k0 + (l>>4)*16 .. +15] -> one 32B load.
static __device__ __forceinline__ v16bf load_b_frag(const __bf16* __restrict__ Wrow,
                                                    int k0, int hi) {
  return *(const v16bf*)(Wrow + k0 + hi * 16);
}

static __device__ __forceinline__ v8f wmma_bf16(v16bf a, v16bf b, v8f c) {
  return __builtin_amdgcn_wmma_f32_16x16x32_bf16(false, a, false, b, (short)0, c,
                                                 false, false);
}

// One K-sweep of D += A @ W^T for a 32(M)x64(N) wave tile:
// 2 A-fragments (rows mbase..+15, mbase+16..+31) x 4 B-fragments, B reused 2x.
static __device__ __forceinline__ void gemm_pass(
    const __bf16* __restrict__ arow0, const __bf16* __restrict__ arow1,
    const __bf16* __restrict__ W, int nlane, int hi, v8f acc[2][4])
{
  for (int k0 = 0; k0 < HID; k0 += 32) {
    v16bf a0 = load_a_frag(arow0, k0, hi);
    v16bf a1 = load_a_frag(arow1, k0, hi);
#pragma unroll
    for (int t = 0; t < 4; ++t) {
      v16bf b = load_b_frag(W + (size_t)(nlane + t * 16) * HID, k0, hi);
      acc[0][t] = wmma_bf16(a0, b, acc[0][t]);
      acc[1][t] = wmma_bf16(a1, b, acc[1][t]);
    }
  }
}

// ---------------- gates = Xbf @ Wih^T + Hbf @ Whh^T + bias  (M=256,N=4096,K=1024)
// Block: 8 waves = 2(M) x 4(N) -> 64M x 256N per block. Wave: 32M x 64N.
__global__ __launch_bounds__(256) void k_gate_gemm(
    const __bf16* __restrict__ Xbf, const __bf16* __restrict__ Hbf,
    const __bf16* __restrict__ Wih, const __bf16* __restrict__ Whh,
    const float* __restrict__ bias, float* __restrict__ gates)
{
  const int lane  = threadIdx.x & 31;
  const int wave  = threadIdx.x >> 5;
  const int mbase = blockIdx.y * 64 + (wave & 1) * 32;
  const int nbase = blockIdx.x * 256 + (wave >> 1) * 64;
  const int r  = lane & 15;
  const int hi = lane >> 4;

  const __bf16* xrow0 = Xbf + (size_t)(mbase + r) * HID;
  const __bf16* xrow1 = xrow0 + (size_t)16 * HID;
  const __bf16* hrow0 = Hbf + (size_t)(mbase + r) * HID;
  const __bf16* hrow1 = hrow0 + (size_t)16 * HID;
  const int nlane = nbase + r;

  v8f acc[2][4] = {};
  gemm_pass(xrow0, xrow1, Wih, nlane, hi, acc);
  gemm_pass(hrow0, hrow1, Whh, nlane, hi, acc);

  // D layout: lane l -> n = l&15 (+tile), m = (l>>4)*8 + v (+16 per sub-tile)
#pragma unroll
  for (int s = 0; s < 2; ++s) {
#pragma unroll
    for (int t = 0; t < 4; ++t) {
      const int n  = nlane + t * 16;
      const float bv = bias[n];
#pragma unroll
      for (int v = 0; v < 8; ++v) {
        const int m = mbase + s * 16 + hi * 8 + v;
        gates[(size_t)m * GATE4 + n] = acc[s][t][v] + bv;
      }
    }
  }
}

// ---------------- LSTM cell elementwise; writes fp32 states to d_out + bf16 feeds
__global__ __launch_bounds__(256) void k_cell(
    const float* __restrict__ gates, float* __restrict__ c_cur,
    float* __restrict__ h_sec, float* __restrict__ c_sec,
    __bf16* __restrict__ Hbf, __bf16* __restrict__ Cbf, int t)
{
  const int idx = blockIdx.x * 256 + threadIdx.x;  // 256*1024 threads
  const int b = idx >> 10;
  const int d = idx & 1023;
  const float* g = gates + (size_t)b * GATE4;
  const float ig = 1.0f / (1.0f + __expf(-g[d]));
  const float fg = 1.0f / (1.0f + __expf(-g[d + HID]));
  const float gg = tanhf(g[d + 2 * HID]);
  const float og = 1.0f / (1.0f + __expf(-g[d + 3 * HID]));
  const float cn = fg * c_cur[idx] + ig * gg;
  const float hn = og * tanhf(cn);
  c_cur[idx] = cn;
  Hbf[idx] = (__bf16)hn;
  Cbf[idx] = (__bf16)cn;
  const size_t o = (size_t)b * (SEQ * HID) + (size_t)t * HID + d;
  h_sec[o] = hn;
  c_sec[o] = cn;
}

// ---------------- o_t = Cbf @ Wfc^T + b_fc ; fp32 -> d_out, bf16 -> next input
__global__ __launch_bounds__(256) void k_fc_gemm(
    const __bf16* __restrict__ Cbf, const __bf16* __restrict__ Wfc,
    const float* __restrict__ bfc,
    float* __restrict__ o_sec, __bf16* __restrict__ Xbf, int t)
{
  const int lane  = threadIdx.x & 31;
  const int wave  = threadIdx.x >> 5;
  const int mbase = blockIdx.y * 64 + (wave & 1) * 32;
  const int nbase = blockIdx.x * 256 + (wave >> 1) * 64;
  const int r  = lane & 15;
  const int hi = lane >> 4;

  const __bf16* crow0 = Cbf + (size_t)(mbase + r) * HID;
  const __bf16* crow1 = crow0 + (size_t)16 * HID;
  const int nlane = nbase + r;

  v8f acc[2][4] = {};
  gemm_pass(crow0, crow1, Wfc, nlane, hi, acc);

#pragma unroll
  for (int s = 0; s < 2; ++s) {
#pragma unroll
    for (int tt = 0; tt < 4; ++tt) {
      const int n  = nlane + tt * 16;
      const float bv = bfc[n];
#pragma unroll
      for (int v = 0; v < 8; ++v) {
        const int m = mbase + s * 16 + hi * 8 + v;
        const float o = acc[s][tt][v] + bv;
        o_sec[(size_t)m * (SEQ * HID) + (size_t)t * HID + n] = o;
        Xbf[(size_t)m * HID + n] = (__bf16)o;   // autoregressive feedback
      }
    }
  }
}

// ---------------- setup kernels ----------------
__global__ void k_f2bf(const float* __restrict__ s, __bf16* __restrict__ d, int n) {
  const int i = blockIdx.x * blockDim.x + threadIdx.x;
  if (i < n) d[i] = (__bf16)s[i];
}
__global__ void k_bias(const float* __restrict__ a, const float* __restrict__ b,
                       float* __restrict__ d, int n) {
  const int i = blockIdx.x * blockDim.x + threadIdx.x;
  if (i < n) d[i] = a[i] + b[i];
}
__global__ void k_init(const float* __restrict__ cvec, __bf16* __restrict__ Xbf,
                       __bf16* __restrict__ Hbf, float* __restrict__ c_cur, int n) {
  const int i = blockIdx.x * blockDim.x + threadIdx.x;
  if (i < n) {
    Xbf[i]   = (__bf16)0.0f;     // i_0 = zeros
    Hbf[i]   = (__bf16)cvec[i];  // h_0 = c_vector
    c_cur[i] = cvec[i];          // c_0 = c_vector
  }
}

extern "C" void kernel_launch(void* const* d_in, const int* in_sizes, int n_in,
                              void* d_out, int out_size, void* d_ws, size_t ws_size,
                              hipStream_t stream)
{
  (void)in_sizes; (void)n_in; (void)out_size; (void)ws_size;
  const float* c_vec = (const float*)d_in[0];
  const float* W_ih  = (const float*)d_in[1];
  const float* W_hh  = (const float*)d_in[2];
  const float* b_ih  = (const float*)d_in[3];
  const float* b_hh  = (const float*)d_in[4];
  const float* W_fc  = (const float*)d_in[5];
  const float* b_fc  = (const float*)d_in[6];

  // workspace carve-out (~25.7 MB total)
  char* ws = (char*)d_ws;
  size_t off = 0;
  auto carve = [&](size_t bytes) -> void* {
    void* p = ws + off;
    off = (off + bytes + 255) & ~(size_t)255;
    return p;
  };
  __bf16* Wih_bf = (__bf16*)carve((size_t)GATE4 * HID * 2);
  __bf16* Whh_bf = (__bf16*)carve((size_t)GATE4 * HID * 2);
  __bf16* Wfc_bf = (__bf16*)carve((size_t)HID * HID * 2);
  float*  bias   = (float*) carve((size_t)GATE4 * 4);
  float*  gates  = (float*) carve((size_t)NBATCH * GATE4 * 4);
  __bf16* Xbf    = (__bf16*)carve((size_t)NBATCH * HID * 2);
  __bf16* Hbf    = (__bf16*)carve((size_t)NBATCH * HID * 2);
  __bf16* Cbf    = (__bf16*)carve((size_t)NBATCH * HID * 2);
  float*  c_cur  = (float*) carve((size_t)NBATCH * HID * 4);

  float* h_sec = (float*)d_out;                          // [256,256,1024]
  float* c_sec = h_sec + (size_t)NBATCH * SEQ * HID;
  float* o_sec = c_sec + (size_t)NBATCH * SEQ * HID;

  const int nW  = GATE4 * HID;      // 4,194,304
  const int nFc = HID * HID;        // 1,048,576
  const int nSt = NBATCH * HID;     // 262,144

  k_f2bf<<<(nW  + 255) / 256, 256, 0, stream>>>(W_ih, Wih_bf, nW);
  k_f2bf<<<(nW  + 255) / 256, 256, 0, stream>>>(W_hh, Whh_bf, nW);
  k_f2bf<<<(nFc + 255) / 256, 256, 0, stream>>>(W_fc, Wfc_bf, nFc);
  k_bias<<<(GATE4 + 255) / 256, 256, 0, stream>>>(b_ih, b_hh, bias, GATE4);
  k_init<<<(nSt + 255) / 256, 256, 0, stream>>>(c_vec, Xbf, Hbf, c_cur, nSt);

  const dim3 gGate(GATE4 / 256, NBATCH / 64);  // 16 x 4 blocks, 8 waves each
  const dim3 gFc  (HID   / 256, NBATCH / 64);  // 4 x 4 blocks

  for (int t = 0; t < SEQ; ++t) {
    k_gate_gemm<<<gGate, 256, 0, stream>>>(Xbf, Hbf, Wih_bf, Whh_bf, bias, gates);
    k_cell<<<nSt / 256, 256, 0, stream>>>(gates, c_cur, h_sec, c_sec, Hbf, Cbf, t);
    k_fc_gemm<<<gFc, 256, 0, stream>>>(Cbf, Wfc_bf, b_fc, o_sec, Xbf, t);
  }
}
